// ImprovedMLPDecoder_84920093376639
// MI455X (gfx1250) — matrix operand order repaired and checked by reading, not compiled, not run on hardware
//
#include <hip/hip_runtime.h>
#include <hip/hip_bf16.h>

typedef __attribute__((ext_vector_type(16))) __bf16 v16bf;
typedef __attribute__((ext_vector_type(8)))  float  v8f;
typedef __attribute__((ext_vector_type(4)))  unsigned int u32x4;
typedef __attribute__((ext_vector_type(8)))  int i32x8;
typedef __attribute__((ext_vector_type(4)))  int i32x4;

union Frag32 { v16bf v; uint4 q[2]; };

__device__ __forceinline__ unsigned short f2bf(float f) {
  __hip_bfloat16 h = __float2bfloat16(f);
  return __builtin_bit_cast(unsigned short, h);
}
__device__ __forceinline__ unsigned pack2(float a, float b) {
  return (unsigned)f2bf(a) | ((unsigned)f2bf(b) << 16);
}
__device__ __forceinline__ float gelu_exact(float x) {
  return 0.5f * x * (1.0f + erff(x * 0.7071067811865475f));
}

// ---- packed weight blob layout (same in d_ws and in LDS) ----
#define OFF_W1 0          // 16kt * 8nt * 32lanes * 32B = 131072
#define OFF_WS 131072     // 16kt * 4nt * 32 * 32      =  65536
#define OFF_W2 196608     //  4kt * 4nt * 32 * 32      =  16384
#define PACK_BYTES 212992 // 8-byte elements: 26624 (fits 16-bit tile_dim0)
// ---- dynamic LDS layout ----
#define OFF_X  212992     // 4 waves * 16*512 bf16     =  65536
#define OFF_H1 278528     // 4 waves * 16*128 bf16     =  16384
#define SMEM_BYTES 294912 // 288 KB of the 320 KB WGP LDS

// f32 [K x N] row-major -> bf16 WMMA B-fragments.
__device__ void pack_wmma_b(const float* __restrict__ W, int K, int N,
                            char* dst, int tid, int nth) {
  const int NT = N >> 4;
  const int total = (K >> 5) * NT * 32;
  for (int id = tid; id < total; id += nth) {
    int kt  = id / (NT * 32);
    int rem = id - kt * NT * 32;
    int l   = rem & 31;
    int n   = ((rem >> 5) << 4) + (l & 15);
    int ks  = (kt << 5) + ((l >> 4) << 4);
    const float* col = W + (size_t)ks * N + n;
    uint4 o0, o1;
    o0.x = pack2(col[0 * N],  col[1 * N]);
    o0.y = pack2(col[2 * N],  col[3 * N]);
    o0.z = pack2(col[4 * N],  col[5 * N]);
    o0.w = pack2(col[6 * N],  col[7 * N]);
    o1.x = pack2(col[8 * N],  col[9 * N]);
    o1.y = pack2(col[10 * N], col[11 * N]);
    o1.z = pack2(col[12 * N], col[13 * N]);
    o1.w = pack2(col[14 * N], col[15 * N]);
    uint4* p = (uint4*)(dst + ((size_t)id << 5));
    p[0] = o0; p[1] = o1;
  }
}

__global__ void pack_weights_kernel(const float* __restrict__ W1,
                                    const float* __restrict__ Ws,
                                    const float* __restrict__ W2,
                                    char* __restrict__ dst) {
  int gid = blockIdx.x * blockDim.x + threadIdx.x;
  int nth = gridDim.x * blockDim.x;
  pack_wmma_b(W1, 512, 128, dst + OFF_W1, gid, nth);
  pack_wmma_b(Ws, 512, 64,  dst + OFF_WS, gid, nth);
  pack_wmma_b(W2, 128, 64,  dst + OFF_W2, gid, nth);
}

__global__ void cvt_tables_kernel(const float* __restrict__ drug,
                                  const float* __restrict__ dis,
                                  char* __restrict__ out,
                                  long long nd, long long ntot) {
  long long i = ((long long)blockIdx.x * blockDim.x + threadIdx.x) * 8;
  long long stride = (long long)gridDim.x * blockDim.x * 8;
  for (; i < ntot; i += stride) {
    const float* p = (i < nd) ? (drug + i) : (dis + (i - nd));
    float4 A = ((const float4*)p)[0];
    float4 B = ((const float4*)p)[1];
    uint4 o;
    o.x = pack2(A.x, A.y); o.y = pack2(A.z, A.w);
    o.z = pack2(B.x, B.y); o.w = pack2(B.z, B.w);
    *(uint4*)(out + i * 2) = o;
  }
}

// TDM: DMA `bytes` (multiple of 8, <=512KB) from global to LDS offset ldsOff.
__device__ __forceinline__ void tdm_copy_to_lds(const void* gaddr,
                                                unsigned ldsOff, unsigned bytes) {
  unsigned long long ga = (unsigned long long)gaddr;
  unsigned n8 = bytes >> 3;
  u32x4 g0;
  g0[0] = 1u;                                    // count=1, no gather
  g0[1] = ldsOff;                                // lds_addr
  g0[2] = (unsigned)ga;                          // global_addr[31:0]
  g0[3] = (unsigned)((ga >> 32) & 0x1FFFFFFull)  // global_addr[56:32]
          | (2u << 30);                          // type=2 ("image")
  i32x8 g1;
  g1[0] = (int)(3u << 16);                       // data_size = 8B
  g1[1] = (int)((n8 & 0xFFFFu) << 16);           // tensor_dim0[15:0]
  g1[2] = (int)(((n8 >> 16) & 0xFFFFu)           // tensor_dim0[31:16]
          | (1u << 16));                         // tensor_dim1 = 1
  g1[3] = (int)((n8 & 0xFFFFu) << 16);           // tile_dim0 = n8
  g1[4] = 0;                                     // tile_dim1/2 unused
  g1[5] = (int)n8;                               // tensor_dim0_stride lo
  g1[6] = 0;
  g1[7] = 0;
  i32x4 z4 = {0, 0, 0, 0};
#if defined(__clang_major__) && (__clang_major__ >= 23)
  i32x8 z8 = {0, 0, 0, 0, 0, 0, 0, 0};
  __builtin_amdgcn_tensor_load_to_lds(g0, g1, z4, z4, z8, 0);
#else
  __builtin_amdgcn_tensor_load_to_lds(g0, g1, z4, z4, 0);
#endif
}

template <bool BF16TAB>
__global__ __launch_bounds__(128)
void fused_edge_mlp(const void* __restrict__ tabDrug, const void* __restrict__ tabDis,
                    const int* __restrict__ src,  const int* __restrict__ dstI,
                    const char* __restrict__ packedW,
                    const float* __restrict__ b1, const float* __restrict__ g1,
                    const float* __restrict__ be1,
                    const float* __restrict__ b2, const float* __restrict__ g2,
                    const float* __restrict__ be2,
                    const float* __restrict__ W3, const float* __restrict__ b3,
                    const float* __restrict__ bs,
                    float* __restrict__ out, int E, int nTiles) {
  extern __shared__ char smem[];
  char* w1f = smem + OFF_W1;
  char* wsf = smem + OFF_WS;
  char* w2f = smem + OFF_W2;

  const int tid = threadIdx.x;
  // Phase 0: wave 0 DMAs the pre-packed bf16 weight fragments into LDS via TDM.
  if ((tid >> 5) == 0) {
    tdm_copy_to_lds(packedW, 0u, PACK_BYTES);
    __builtin_amdgcn_s_wait_tensorcnt(0);
  }
  __syncthreads();

  const int wv   = tid >> 5;
  const int lane = tid & 31;
  const int tile = blockIdx.x * 4 + wv;
  if (tile >= nTiles) return;
  const long long e0 = (long long)tile << 4;

  char* xt = smem + OFF_X  + (wv << 14);  // [16][512] bf16, row stride 1024 B
  char* h1 = smem + OFF_H1 + (wv << 12);  // [16][128] bf16, row stride 256 B

  // Phase 1: gather x = [drug[src] || dis[dst]] into LDS as bf16
  for (int idx = lane; idx < 1024; idx += 32) {
    int m = idx >> 6, c = idx & 63;
    long long e = e0 + m; if (e >= E) e = E - 1;
    int f = c << 3;
    uint4 o;
    if (BF16TAB) {
      const char* row = (f < 256)
          ? ((const char*)tabDrug + ((size_t)src[e]  * 512 + (size_t)f * 2))
          : ((const char*)tabDis  + ((size_t)dstI[e] * 512 + (size_t)(f - 256) * 2));
      o = *(const uint4*)row;
    } else {
      const float* row = (f < 256)
          ? ((const float*)tabDrug + (size_t)src[e]  * 256 + f)
          : ((const float*)tabDis  + (size_t)dstI[e] * 256 + (f - 256));
      float4 A = ((const float4*)row)[0];
      float4 B = ((const float4*)row)[1];
      o.x = pack2(A.x, A.y); o.y = pack2(A.z, A.w);
      o.z = pack2(B.x, B.y); o.w = pack2(B.z, B.w);
    }
    *(uint4*)(xt + m * 1024 + c * 16) = o;
  }

  const int ln = lane & 15;   // column / row-in-tile
  const int kh = lane >> 4;   // K-half for A, row-half for C
  const v8f vzero = {0, 0, 0, 0, 0, 0, 0, 0};

  // A fragment from a row-major bf16 tile
  auto ldA = [&](const char* t, int rowStride, int kt) -> Frag32 {
    Frag32 a;
    const char* ap = t + ln * rowStride + kt * 64 + kh * 32;
    a.q[0] = *(const uint4*)(ap);
    a.q[1] = *(const uint4*)(ap + 16);
    return a;
  };
  // Combined B operand [W1 | Ws] : nt 0..7 -> W1, nt 8..11 -> Ws
  auto ldB1 = [&](int kt, int nt) -> Frag32 {
    const char* base = (nt < 8)
        ? (w1f + (((kt * 8 + nt)       * 32 + lane) << 5))
        : (wsf + (((kt * 4 + (nt - 8)) * 32 + lane) << 5));
    Frag32 b;
    const uint4* bp = (const uint4*)base;
    b.q[0] = bp[0]; b.q[1] = bp[1];
    return b;
  };

  // ---- fused pass: [16x512] @ [512x(128|64)] -> h1pre (nt 0..7), identity (nt 8..11)
  // one-deep software pipeline on both A and B fragments
  v8f acc[12];
  #pragma unroll
  for (int nt = 0; nt < 12; nt++) acc[nt] = vzero;
  {
    Frag32 a  = ldA(xt, 1024, 0);
    Frag32 b  = ldB1(0, 0);
    #pragma unroll
    for (int kt = 0; kt < 16; kt++) {
      Frag32 an = (kt < 15) ? ldA(xt, 1024, kt + 1) : a;
      #pragma unroll
      for (int nt = 0; nt < 12; nt++) {
        Frag32 bn = (nt < 11) ? ldB1(kt, nt + 1)
                  : ((kt < 15) ? ldB1(kt + 1, 0) : b);
        acc[nt] = __builtin_amdgcn_wmma_f32_16x16x32_bf16(
            false, a.v, false, b.v, (short)0, acc[nt], false, false);
        b = bn;
      }
      a = an;
    }
  }

  // bias + LayerNorm(128) + exact GELU on nt 0..7, write h1 (bf16) to LDS
  float b1v[8], g1v[8], bev[8];
  #pragma unroll
  for (int nt = 0; nt < 8; nt++) {
    b1v[nt] = b1[nt * 16 + ln];
    g1v[nt] = g1[nt * 16 + ln];
    bev[nt] = be1[nt * 16 + ln];
  }
  #pragma unroll
  for (int nt = 0; nt < 8; nt++)
    #pragma unroll
    for (int v = 0; v < 8; v++) acc[nt][v] += b1v[nt];

  #pragma unroll
  for (int v = 0; v < 8; v++) {
    float s = 0.f, q = 0.f;
    #pragma unroll
    for (int nt = 0; nt < 8; nt++) { float x = acc[nt][v]; s += x; q += x * x; }
    #pragma unroll
    for (int msk = 1; msk < 16; msk <<= 1) {   // 16-lane half-wave reduction
      s += __shfl_xor(s, msk, 32);
      q += __shfl_xor(q, msk, 32);
    }
    float mu  = s * (1.f / 128.f);
    float var = q * (1.f / 128.f) - mu * mu;
    float inv = rsqrtf(var + 1e-5f);
    int row = v + (kh << 3);
    #pragma unroll
    for (int nt = 0; nt < 8; nt++) {
      float y = (acc[nt][v] - mu) * inv * g1v[nt] + bev[nt];
      y = gelu_exact(y);
      *(unsigned short*)(h1 + (row * 128 + nt * 16 + ln) * 2) = f2bf(y);
    }
  }

  // ---- fc2: [16x128] @ [128x64], software-pipelined ----
  v8f h2[4];
  #pragma unroll
  for (int nt = 0; nt < 4; nt++) h2[nt] = vzero;
  {
    auto ldB2 = [&](int kt, int nt) -> Frag32 {
      Frag32 b;
      const uint4* bp = (const uint4*)(w2f + (((kt * 4 + nt) * 32 + lane) << 5));
      b.q[0] = bp[0]; b.q[1] = bp[1];
      return b;
    };
    Frag32 a = ldA(h1, 256, 0);
    Frag32 b = ldB2(0, 0);
    #pragma unroll
    for (int kt = 0; kt < 4; kt++) {
      Frag32 an = (kt < 3) ? ldA(h1, 256, kt + 1) : a;
      #pragma unroll
      for (int nt = 0; nt < 4; nt++) {
        Frag32 bn = (nt < 3) ? ldB2(kt, nt + 1)
                  : ((kt < 3) ? ldB2(kt + 1, 0) : b);
        h2[nt] = __builtin_amdgcn_wmma_f32_16x16x32_bf16(
            false, a.v, false, b.v, (short)0, h2[nt], false, false);
        b = bn;
      }
      a = an;
    }
  }

  float bsv[4], b2v[4], g2v[4], be2v[4], w3v[4];
  #pragma unroll
  for (int nt = 0; nt < 4; nt++) {
    bsv[nt]  = bs[nt * 16 + ln];
    b2v[nt]  = b2[nt * 16 + ln];
    g2v[nt]  = g2[nt * 16 + ln];
    be2v[nt] = be2[nt * 16 + ln];
    w3v[nt]  = W3[nt * 16 + ln];
  }
  const float b3v = b3[0];

  // LN(64) + residual (acc[8..11] = identity) + GELU + fc3 dot + store
  #pragma unroll
  for (int v = 0; v < 8; v++) {
    float s = 0.f, q = 0.f;
    #pragma unroll
    for (int nt = 0; nt < 4; nt++) {
      float x = h2[nt][v] + b2v[nt];
      s += x; q += x * x;
    }
    #pragma unroll
    for (int msk = 1; msk < 16; msk <<= 1) {
      s += __shfl_xor(s, msk, 32);
      q += __shfl_xor(q, msk, 32);
    }
    float mu  = s * (1.f / 64.f);
    float var = q * (1.f / 64.f) - mu * mu;
    float inv = rsqrtf(var + 1e-5f);
    float r = 0.f;
    #pragma unroll
    for (int nt = 0; nt < 4; nt++) {
      float y = (h2[nt][v] + b2v[nt] - mu) * inv * g2v[nt] + be2v[nt];
      y += acc[8 + nt][v] + bsv[nt];
      y = gelu_exact(y);
      r += y * w3v[nt];
    }
    #pragma unroll
    for (int msk = 1; msk < 16; msk <<= 1) r += __shfl_xor(r, msk, 32);
    if (ln == 0) {
      long long e = e0 + v + (kh << 3);
      if (e < E) out[e] = r + b3v;
    }
  }
}

extern "C" void kernel_launch(void* const* d_in, const int* in_sizes, int n_in,
                              void* d_out, int out_size, void* d_ws, size_t ws_size,
                              hipStream_t stream) {
  (void)n_in; (void)out_size;
  const float* drug = (const float*)d_in[0];
  const float* dis  = (const float*)d_in[1];
  const int*   src  = (const int*)d_in[2];
  const int*   dst  = (const int*)d_in[3];
  const float* W1  = (const float*)d_in[4];
  const float* b1  = (const float*)d_in[5];
  const float* g1  = (const float*)d_in[6];
  const float* be1 = (const float*)d_in[7];
  const float* W2  = (const float*)d_in[8];
  const float* b2  = (const float*)d_in[9];
  const float* g2  = (const float*)d_in[10];
  const float* be2 = (const float*)d_in[11];
  const float* W3  = (const float*)d_in[12];
  const float* b3  = (const float*)d_in[13];
  const float* Ws  = (const float*)d_in[14];
  const float* bs  = (const float*)d_in[15];
  float* out = (float*)d_out;
  char* wsc = (char*)d_ws;

  int E = in_sizes[2];
  int nTiles  = (E + 15) / 16;
  int nBlocks = (nTiles + 3) / 4;

  // Pre-pass 1: pack weights (bf16 WMMA fragments) into workspace, once.
  pack_weights_kernel<<<26, 256, 0, stream>>>(W1, Ws, W2, wsc);

  // Pre-pass 2 (if workspace allows): bf16 copies of the feature tables.
  long long nd = in_sizes[0], ni = in_sizes[1];
  size_t tabBytes = (size_t)(nd + ni) * 2;
  bool bf16Tab = ws_size >= (size_t)PACK_BYTES + tabBytes;

  if (bf16Tab) {
    char* tab = wsc + PACK_BYTES;   // 16B-aligned (212992 % 16 == 0)
    cvt_tables_kernel<<<2048, 256, 0, stream>>>(drug, dis, tab, nd, nd + ni);
    fused_edge_mlp<true><<<nBlocks, 128, SMEM_BYTES, stream>>>(
        tab, tab + (size_t)nd * 2, src, dst, wsc,
        b1, g1, be1, b2, g2, be2, W3, b3, bs, out, E, nTiles);
  } else {
    fused_edge_mlp<false><<<nBlocks, 128, SMEM_BYTES, stream>>>(
        drug, dis, src, dst, wsc,
        b1, g1, be1, b2, g2, be2, W3, b3, bs, out, E, nTiles);
  }
}
